// TFNLiteLayer_2302102471547
// MI455X (gfx1250) — matrix operand order repaired and checked by reading, not compiled, not run on hardware
//
#include <hip/hip_runtime.h>
#include <hip/hip_bf16.h>
#include <math.h>

typedef __attribute__((ext_vector_type(16))) _Float16 v16h;
typedef __attribute__((ext_vector_type(8)))  _Float16 v8h;
typedef __attribute__((ext_vector_type(8)))  float    v8f;

#define TILE_E   32           // edges per block (2 row tiles of 16)
#define NTHREADS 256          // 8 waves of 32
#define KDIM     64
#define NRED     320          // reduced GEMM width: 4*64 paired paths + 64 (wE)
#define NCT      20           // 320 / 16 column tiles
#define CT_PER_WAVE 5         // 20 tiles / 4 waves per row tile

// ---------------- constants ----------------
#define S15_C  3.8729833462f
#define S5_C   2.2360679775f
#define Q10    0.3162277660f   // 1/sqrt(10)
#define R30_2  0.3651483717f   // 2/sqrt(30)
#define R30_1  0.1825741858f   // 1/sqrt(30)
#define A0_C   0.1767766953f   // 1/sqrt(32)
#define A1_C   0.2738612788f   // sqrt(3/40)
#define INV_S3 0.5773502692f   // 1/sqrt(3)

#define W2P_HALVES (NCT * 2 * 32 * 16)   // 20480 halves = 40960 B
#define W1P_HALVES (4 * 32 * 16)         // 2048  halves = 4096  B

// ---------------------------------------------------------------------------
// Prep A: v-pair-reduced, path-scaled W2 (f32 [64][576] -> f16 [64][320])
// in WMMA B-fragment lane layout (B 32x16 K-major: lane l holds column
// n = ct*16 + (l&15); half hidx holds k = step*32 + (hidx&7) + 8*(l>>4) + 16*(hidx>>3)).
// Reduced columns: [0,64)=A-pairs, [64,128)=B-pairs, [128,192)=C-pairs,
// [192,256)=D-pairs, [256,320)=wE.  Path scales folded in.
// ---------------------------------------------------------------------------
__global__ void tfn_pack_w2(const float* __restrict__ W2, _Float16* __restrict__ w2p) {
    int tid = blockIdx.x * blockDim.x + threadIdx.x;
    if (tid >= W2P_HALVES) return;
    int hidx = tid & 15;
    int l    = (tid >> 4) & 31;
    int step = (tid >> 9) & 1;
    int ct   = tid >> 10;
    int nr = ct * 16 + (l & 15);                 // reduced column 0..319
    int k  = step * 32 + (hidx & 7) + 8 * (l >> 4) + 16 * (hidx >> 3);   // 0..63

    int p   = nr >> 6;        // path 0..4
    int idx = nr & 63;
    int u   = idx >> 3;
    int wch = idx & 7;
    float val, scale;
    if (p < 4) {
        int src = p * 128 + (2 * u) * 8 + wch;   // v=0 column; v=1 is +8
        val = W2[k * 576 + src] + W2[k * 576 + src + 8];
        scale = (p < 2) ? (0.125f * A0_C)
              : (p == 2 ? (0.125f * A1_C)
                        : (0.125f * A1_C * INV_S3));
    } else {
        val = W2[k * 576 + 512 + u * 8 + wch];
        scale = 0.125f * A1_C;
    }
    w2p[tid] = (_Float16)(val * scale);
}

// ---------------------------------------------------------------------------
// Prep B: W1 (f32 [16][64]) -> f16 B-fragments, K padded 16->32 with zeros,
// 1/sqrt(16)=0.25 scale folded.  4 column tiles of 16.
// ---------------------------------------------------------------------------
__global__ void tfn_pack_w1(const float* __restrict__ W1, _Float16* __restrict__ w1p) {
    int tid = blockIdx.x * blockDim.x + threadIdx.x;
    if (tid >= W1P_HALVES) return;
    int hidx = tid & 15;
    int l    = (tid >> 4) & 31;
    int ct   = tid >> 9;                 // 0..3
    int k = (hidx & 7) + 8 * (l >> 4) + 16 * (hidx >> 3);        // 0..31 (16..31 are zero pad)
    int n = ct * 16 + (l & 15);          // 0..63
    w1p[tid] = (k < 16) ? (_Float16)(W1[k * 64 + n] * 0.25f) : (_Float16)0.0f;
}

__global__ void tfn_zero(float* __restrict__ y, int n) {
    int tid = blockIdx.x * blockDim.x + threadIdx.x;
    if (tid < n) y[tid] = 0.0f;
}

__global__ void tfn_finalize(float* __restrict__ y, int n) {
    int tid = blockIdx.x * blockDim.x + threadIdx.x;
    if (tid >= n) return;
    if ((tid & 31) < 8) {
        float v = y[tid];
        y[tid] = v / (1.0f + __expf(-v));   // silu
    }
}

// ---------------------------------------------------------------------------
// Main edge kernel: 32 edges / block, 256 threads (8 waves).
// ---------------------------------------------------------------------------
__global__ void __launch_bounds__(NTHREADS)
tfn_edge_kernel(const float* __restrict__ x,
                const float* __restrict__ pos,
                const int*   __restrict__ ei,
                const _Float16* __restrict__ w1p,
                const _Float16* __restrict__ w2p,
                float* __restrict__ yout,
                int E) {
    __shared__ float                  wshm[TILE_E * NRED];   // 40960 B (reduced w)
    __shared__ __align__(32) _Float16 hshm[TILE_E * KDIM];   // 4096 B
    __shared__ __align__(32) _Float16 rbfh[TILE_E * 32];     // 2048 B (K padded to 32)
    __shared__ float                  nsh[TILE_E][3];
    __shared__ float                  Msh[TILE_E][9];        // sum_j Y2[j]*cg[k,i,j]
    __shared__ int                    rowsh[TILE_E];
    __shared__ float                  xrow[TILE_E][32];

    const int t    = threadIdx.x;
    const int base = blockIdx.x * TILE_E;

    const int wave   = t >> 5;            // 0..7
    const int lane   = t & 31;
    const int laneHi = lane >> 4;
    const int koff   = laneHi * 8;        // A-frag K offset for hi half-wave
    const int m      = lane & 15;
    const int rt     = wave >> 2;         // row tile 0..1 (16 edges each)

    // ---- phase 1: geometry + f16 RBF (one thread per edge) ----
    if (t < TILE_E) {
        const int eg = base + t;
        int row = 0, col = 0;
        if (eg < E) { row = ei[eg]; col = ei[E + eg]; }
        rowsh[t] = row;
        float px = 0.f, py = 0.f, pz = 0.f;
        if (eg < E) {
            px = pos[row * 3 + 0] - pos[col * 3 + 0];
            py = pos[row * 3 + 1] - pos[col * 3 + 1];
            pz = pos[row * 3 + 2] - pos[col * 3 + 2];
        }
        float d   = sqrtf(px * px + py * py + pz * pz + 1e-12f);
        float inv = 1.0f / d;
        float nx = px * inv, ny = py * inv, nz = pz * inv;
        nsh[t][0] = nx; nsh[t][1] = ny; nsh[t][2] = nz;
        // Y2 and cg-folded symmetric 3x3: M[k][i] = sum_j Y2[j]*cg[k,i,j]
        float y20 = S15_C * nx * nz;
        float y21 = S15_C * nx * ny;
        float y22 = 0.5f * S5_C * (3.0f * ny * ny - 1.0f);
        float y23 = S15_C * ny * nz;
        float y24 = 0.5f * S15_C * (nz * nz - nx * nx);
        float m01 = y21 * Q10, m02 = y20 * Q10, m12 = y23 * Q10;
        Msh[t][0] = -y22 * R30_1 - y24 * Q10;   // M00
        Msh[t][1] = m01;  Msh[t][2] = m02;
        Msh[t][3] = m01;
        Msh[t][4] = y22 * R30_2;                // M11
        Msh[t][5] = m12;
        Msh[t][6] = m02;  Msh[t][7] = m12;
        Msh[t][8] = -y22 * R30_1 + y24 * Q10;   // M22
        // 16 Gaussian RBFs (f16), mu_k = k/3, 1/(2*delta^2) = 4.5; pad K to 32
        #pragma unroll
        for (int k = 0; k < 16; ++k) {
            float diff = d - (float)k * (1.0f / 3.0f);
            float v = (eg < E) ? __expf(-diff * diff * 4.5f) : 0.0f;
            rbfh[t * 32 + k] = (_Float16)v;
            rbfh[t * 32 + 16 + k] = (_Float16)0.0f;
        }
    }
    __syncthreads();

    // ---- phase 2: gather x rows + WMMA  h = relu(rbf @ W1/4)  (32x64, K=32) ----
    {
        // x-row gather (8 threads per edge, 4 floats each)
        const int e = t >> 3;
        const int q = t & 7;
        const int row = rowsh[e];
        #pragma unroll
        for (int j = 0; j < 4; ++j)
            xrow[e][q * 4 + j] = x[row * 32 + q * 4 + j];

        // one 16x16 h-tile per wave: rt = row tile, ctn = W1 column tile 0..3
        const int ctn = wave & 3;
        const _Float16* rp = &rbfh[(rt * 16 + m) * 32 + koff];
        v8h lo = *(const v8h*)rp;
        v8h hi = *(const v8h*)(rp + 16);
        v16h a;
        #pragma unroll
        for (int i = 0; i < 8; ++i) { a[i] = lo[i]; a[8 + i] = hi[i]; }
        const v16h b = *(const v16h*)(w1p + ((size_t)ctn * 32 + lane) * 16);
        v8f hacc = {};
        hacc = __builtin_amdgcn_wmma_f32_16x16x32_f16(
            false, a, false, b, (short)0, hacc, false, false);
        // relu + f16 convert, scatter into hshm[edge][j]
        #pragma unroll
        for (int r = 0; r < 8; ++r) {
            const int edge = rt * 16 + r + laneHi * 8;
            hshm[edge * KDIM + ctn * 16 + m] = (_Float16)fmaxf(hacc[r], 0.0f);
        }
    }
    __syncthreads();

    // ---- phase 3: WMMA  w_red = h @ W2_red  (16 x 320 per row tile, K = 64) ----
    {
        const int ctbase = (wave & 3) * CT_PER_WAVE;

        // A fragments depend only on step -> hoist (4 x 16B LDS reads total)
        v16h afrag[2];
        #pragma unroll
        for (int step = 0; step < 2; ++step) {
            const _Float16* hp = &hshm[(rt * 16 + m) * KDIM + step * 32 + koff];
            v8h lo = *(const v8h*)hp;
            v8h hi = *(const v8h*)(hp + 16);
            #pragma unroll
            for (int i = 0; i < 8; ++i) { afrag[step][i] = lo[i]; afrag[step][8 + i] = hi[i]; }
        }

        #pragma unroll
        for (int j = 0; j < CT_PER_WAVE; ++j) {
            const int ct = ctbase + j;        // 0..19
            v8f acc = {};
            #pragma unroll
            for (int step = 0; step < 2; ++step) {
                const v16h b =
                    *(const v16h*)(w2p + ((size_t)(ct * 2 + step) * 32 + lane) * 16);
                acc = __builtin_amdgcn_wmma_f32_16x16x32_f16(
                    false, afrag[step], false, b, (short)0, acc, false, false);
            }
            const int colg = ct * 16 + m;
            #pragma unroll
            for (int r = 0; r < 8; ++r)
                wshm[(rt * 16 + r + laneHi * 8) * NRED + colg] = acc[r];
        }
    }
    __syncthreads();

    // ---- phase 4: per-edge equivariant contraction + atomic scatter ----
    {
        const int e = t >> 3;        // edge in tile
        const int c = t & 7;         // output channel 0..7
        const int eg = base + e;
        if (eg < E) {
            const float* wl = &wshm[e * NRED];
            const float nx = nsh[e][0], ny = nsh[e][1], nz = nsh[e][2];
            float Ms[9];
            #pragma unroll
            for (int i = 0; i < 9; ++i) Ms[i] = Msh[e][i];

            float acc_s = 0.f, av0 = 0.f, av1 = 0.f, av2 = 0.f;
            #pragma unroll
            for (int u = 0; u < 8; ++u) {
                const float xsu = xrow[e][u];
                const float x0 = xrow[e][8 + 3 * u + 0];
                const float x1 = xrow[e][8 + 3 * u + 1];
                const float x2 = xrow[e][8 + 3 * u + 2];
                const float dot = x0 * nx + x1 * ny + x2 * nz;
                const int uo = u * 8 + c;
                const float wa = wl[uo];             // A-pairs  (a0/8 folded)
                const float wb = wl[64  + uo];       // B-pairs  (a0/8)
                const float wc = wl[128 + uo];       // C-pairs  (a1/8)
                const float wd = wl[192 + uo];       // D-pairs  (a1/(8*sqrt3))
                const float we = wl[256 + uo];       // wE       (a1/8)
                acc_s = fmaf(wa, xsu, fmaf(wb, dot, acc_s));
                const float t0 = Ms[0] * x0 + Ms[1] * x1 + Ms[2] * x2;
                const float t1 = Ms[3] * x0 + Ms[4] * x1 + Ms[5] * x2;
                const float t2 = Ms[6] * x0 + Ms[7] * x1 + Ms[8] * x2;
                const float cs = wc * xsu;
                av0 = fmaf(cs, nx, fmaf(wd, x0, fmaf(we, t0, av0)));
                av1 = fmaf(cs, ny, fmaf(wd, x1, fmaf(we, t1, av1)));
                av2 = fmaf(cs, nz, fmaf(wd, x2, fmaf(we, t2, av2)));
            }
            float* yb = yout + (size_t)rowsh[e] * 32;
            atomicAdd(yb + c, acc_s);
            atomicAdd(yb + 8 + c * 3 + 0, av0);
            atomicAdd(yb + 8 + c * 3 + 1, av1);
            atomicAdd(yb + 8 + c * 3 + 2, av2);
        }
    }
}

extern "C" void kernel_launch(void* const* d_in, const int* in_sizes, int n_in,
                              void* d_out, int out_size, void* d_ws, size_t ws_size,
                              hipStream_t stream) {
    const float* x   = (const float*)d_in[0];
    const float* pos = (const float*)d_in[1];
    const int*   ei  = (const int*)d_in[2];
    const float* W1  = (const float*)d_in[3];
    const float* W2  = (const float*)d_in[4];
    float* yout = (float*)d_out;

    const int N = in_sizes[0] / 32;
    const int E = in_sizes[2] / 2;

    _Float16* w2p = (_Float16*)d_ws;                 // 40960 B
    _Float16* w1p = (_Float16*)d_ws + W2P_HALVES;    // 4096 B (32B aligned)

    tfn_pack_w2<<<(W2P_HALVES + 255) / 256, 256, 0, stream>>>(W2, w2p);
    tfn_pack_w1<<<(W1P_HALVES + 255) / 256, 256, 0, stream>>>(W1, w1p);
    tfn_zero<<<(N * 32 + 255) / 256, 256, 0, stream>>>(yout, N * 32);

    const int nb = (E + TILE_E - 1) / TILE_E;
    tfn_edge_kernel<<<nb, NTHREADS, 0, stream>>>(x, pos, ei, w1p, w2p, yout, E);

    tfn_finalize<<<(N * 32 + 255) / 256, 256, 0, stream>>>(yout, N * 32);
}
